// LorentzDynamics_87668872446026
// MI455X (gfx1250) — compile-verified, matrix operand order checked by reading, not run on hardware
//
#include <hip/hip_runtime.h>
#include <cstdint>
#include <cstddef>

// ---------------------------------------------------------------------------
// LorentzDynamics for MI455X (gfx1250, wave32, WMMA)
// bf16 WMMA GEMMs (f32 accumulate) with triple-buffered
// GLOBAL_LOAD_ASYNC_TO_LDS staging (s_wait_asynccnt 4 pipelining),
// plus fused hyperbolic row-map kernels.
// ---------------------------------------------------------------------------

typedef __attribute__((ext_vector_type(16))) __bf16 v16bf;
typedef __attribute__((ext_vector_type(8)))  __bf16 v8bf;
typedef __attribute__((ext_vector_type(8)))  float  v8f;

#define EPSF  1e-7f
#define MAXN  10.0f

__device__ __forceinline__ float sinh_div(float d) {
  float ds = fmaxf(d, 1e-6f);
  return (d < 1e-6f) ? 1.0f : sinhf(ds) / ds;
}
__device__ __forceinline__ float div_sinh(float d) {
  float ds = fmaxf(d, 1e-6f);
  return (d < 1e-6f) ? 1.0f : ds / sinhf(ds);
}

// 256-thread block reduction (8 wave32s)
__device__ __forceinline__ float block_reduce256(float v, float* sm) {
  const int t = threadIdx.x;
  sm[t] = v;
  __syncthreads();
  #pragma unroll
  for (int s = 128; s > 0; s >>= 1) {
    if (t < s) sm[t] += sm[t + s];
    __syncthreads();
  }
  float r = sm[0];
  __syncthreads();
  return r;
}

// --- gfx1250 async memory->LDS copy (ASYNCcnt-tracked) ---------------------
__device__ __forceinline__ void async_b128(uint32_t lds_off, uint64_t gaddr) {
  asm volatile("global_load_async_to_lds_b128 %0, %1, off"
               :: "v"(lds_off), "v"(gaddr)
               : "memory");
}
__device__ __forceinline__ void wait_async0() {
  asm volatile("s_wait_asynccnt 0x0" ::: "memory");
}
// allow one tile-stage (4 async ops per wave) to remain in flight
__device__ __forceinline__ void wait_async4() {
  asm volatile("s_wait_asynccnt 0x4" ::: "memory");
}

// ---------------------------------------------------------------------------
// f32 -> bf16 weight conversion
// ---------------------------------------------------------------------------
__global__ __launch_bounds__(256)
void f32_to_bf16(const float* __restrict__ src, __bf16* __restrict__ dst, int n) {
  int i = blockIdx.x * 256 + threadIdx.x;
  if (i < n) dst[i] = (__bf16)src[i];
}

// ---------------------------------------------------------------------------
// Prep: A0 = logmap0(projx(safe_expmap0(logmap0(z_h))))  -> bf16 (N, 256)
// ---------------------------------------------------------------------------
__global__ __launch_bounds__(256)
void prep_rows(const float* __restrict__ zh, __bf16* __restrict__ A0) {
  __shared__ float sm[256];
  const size_t row = blockIdx.x;
  const int j = threadIdx.x;
  const float z  = zh[row * 256 + j];
  const float x0 = fmaxf(zh[row * 256], 1.0f + EPSF);
  const float d  = acoshf(x0);
  const float t  = (j == 0) ? 0.0f : div_sinh(d) * z;   // tangent, t[0]=0
  const float S  = block_reduce256(t * t, sm);
  const float n  = sqrtf(S);
  const float sc = fminf(1.0f, MAXN / fmaxf(n, EPSF));
  const float m  = fminf(n, MAXN);
  const float sp = sinh_div(m) * sc * t;                // spatial of v_hyp
  const float d2 = acoshf(fmaxf(coshf(m), 1.0f + EPSF));
  const float a  = (j == 0) ? 0.0f : div_sinh(d2) * sp; // logmap0(v_hyp)
  A0[row * 256 + j] = (__bf16)a;
}

// ---------------------------------------------------------------------------
// WMMA GEMM:  C(NxHout) = A(NxK) @ W(HoutxK)^T + bias,  A/W bf16, C f32
// 256 threads = 8 waves; BM=128, BN=128, BK=32; triple-buffered LDS staged
// with global_load_async_to_lds_b128 (waits relaxed to asynccnt<=4 so one
// tile copy is always in flight). Each wave: 2x4 grid of
// v_wmma_f32_16x16x32_bf16 (32x64 output patch).
// ---------------------------------------------------------------------------
#define BM   128
#define BN   128
#define BK   32
#define LDAH 40   // 32 + 8 halves pad (16B) to stagger LDS banks
#define NBUF 3

__global__ __launch_bounds__(256)
void gemm_bf16_wmma(const __bf16* __restrict__ A, const __bf16* __restrict__ W,
                    const float* __restrict__ bias, float* __restrict__ C,
                    int K, int Hout) {
  __shared__ __bf16 As[NBUF][BM * LDAH];
  __shared__ __bf16 Ws[NBUF][BN * LDAH];

  const int tid  = threadIdx.x;
  const int wave = tid >> 5, lane = tid & 31;
  const int waveM = wave & 3;      // 4 waves along M (32 rows each)
  const int waveN = wave >> 2;     // 2 waves along N (64 cols each)
  const int r    = lane & 15;
  const int ksel = lane >> 4;      // 0: K chunks [0,8)+[16,24); 1: [8,16)+[24,32)

  const size_t rowBase = (size_t)blockIdx.x * BM;
  const int    colBase = blockIdx.y * BN;

  // staging: both A and W blocks are 128 rows x 32 halves; each thread owns a
  // 32B (two-b128) chunk of one row in each.
  const int srow = tid >> 1;
  const int soff = (tid & 1) * 16;   // halves

  const __bf16* aG = A + (rowBase + (size_t)srow) * (size_t)K + soff;
  const __bf16* wG = W + ((size_t)(colBase + srow)) * (size_t)K + soff;

  uint32_t aL[NBUF], wL[NBUF];
  #pragma unroll
  for (int b = 0; b < NBUF; b++) {
    aL[b] = (uint32_t)(uintptr_t)(&As[b][srow * LDAH + soff]);
    wL[b] = (uint32_t)(uintptr_t)(&Ws[b][srow * LDAH + soff]);
  }

  const int KT = K / BK;
  v8f acc[2][4] = {};

  // prologue: async-stage tiles 0 and 1 into buffers 0 and 1
  {
    async_b128(aL[0],      (uint64_t)(uintptr_t)(aG));
    async_b128(aL[0] + 16, (uint64_t)(uintptr_t)(aG + 8));
    async_b128(wL[0],      (uint64_t)(uintptr_t)(wG));
    async_b128(wL[0] + 16, (uint64_t)(uintptr_t)(wG + 8));
  }
  if (KT > 1) {
    const __bf16* ag = aG + BK;
    const __bf16* wg = wG + BK;
    async_b128(aL[1],      (uint64_t)(uintptr_t)(ag));
    async_b128(aL[1] + 16, (uint64_t)(uintptr_t)(ag + 8));
    async_b128(wL[1],      (uint64_t)(uintptr_t)(wg));
    async_b128(wL[1] + 16, (uint64_t)(uintptr_t)(wg + 8));
    wait_async4();   // tile 0 landed; tile 1 still in flight
  } else {
    wait_async0();
  }
  __syncthreads();

  int cur = 0;                        // LDS buffer holding tile kt
  for (int kt = 0; kt < KT; ++kt) {
    const bool hasNext  = (kt + 1) < KT;
    const bool hasNext2 = (kt + 2) < KT;

    if (hasNext2) {
      int nxt2 = cur + 2; if (nxt2 >= NBUF) nxt2 -= NBUF;
      const __bf16* ag = aG + (size_t)(kt + 2) * BK;
      const __bf16* wg = wG + (size_t)(kt + 2) * BK;
      async_b128(aL[nxt2],      (uint64_t)(uintptr_t)(ag));
      async_b128(aL[nxt2] + 16, (uint64_t)(uintptr_t)(ag + 8));
      async_b128(wL[nxt2],      (uint64_t)(uintptr_t)(wg));
      async_b128(wL[nxt2] + 16, (uint64_t)(uintptr_t)(wg + 8));
    }

    // fragment loads from LDS (16-bit A 16x32 layout; B symmetric, W row-major)
    v16bf afr[2], wfr[4];
    #pragma unroll
    for (int i = 0; i < 2; i++) {
      const __bf16* p = &As[cur][(waveM * 32 + i * 16 + r) * LDAH];
      v8bf lo = *(const v8bf*)(p + ksel * 8);
      v8bf hi = *(const v8bf*)(p + 16 + ksel * 8);
      #pragma unroll
      for (int e = 0; e < 8; e++) { afr[i][e] = lo[e]; afr[i][8 + e] = hi[e]; }
    }
    #pragma unroll
    for (int j = 0; j < 4; j++) {
      const __bf16* p = &Ws[cur][(waveN * 64 + j * 16 + r) * LDAH];
      v8bf lo = *(const v8bf*)(p + ksel * 8);
      v8bf hi = *(const v8bf*)(p + 16 + ksel * 8);
      #pragma unroll
      for (int e = 0; e < 8; e++) { wfr[j][e] = lo[e]; wfr[j][8 + e] = hi[e]; }
    }

    #pragma unroll
    for (int i = 0; i < 2; i++)
      #pragma unroll
      for (int j = 0; j < 4; j++)
        acc[i][j] = __builtin_amdgcn_wmma_f32_16x16x32_bf16(
            false, afr[i], false, wfr[j], (short)0, acc[i][j], false, false);

    if (hasNext) {
      // require tile kt+1's copy done; tile kt+2's (4 async ops) may remain
      if (hasNext2) wait_async4(); else wait_async0();
      __syncthreads();
    }

    cur = (cur + 1 == NBUF) ? 0 : cur + 1;
  }

  // epilogue: C/D layout — lanes 0-15: M=e, N=lane; lanes 16-31: M=8+e, N=lane-16
  #pragma unroll
  for (int i = 0; i < 2; i++) {
    const size_t row0 = rowBase + waveM * 32 + i * 16 + ksel * 8;
    #pragma unroll
    for (int j = 0; j < 4; j++) {
      const int   col = colBase + waveN * 64 + j * 16 + r;
      const float b   = bias[col];
      #pragma unroll
      for (int e = 0; e < 8; e++)
        C[(row0 + e) * (size_t)Hout + col] = acc[i][j][e] + b;
    }
  }
}

// ---------------------------------------------------------------------------
// Layer epilogue: Y(N,H) f32 -> A_next(N,H) bf16
// A_next = logmap0(projx(safe_expmap0(Y)))  (inter-layer renorm is identity
// because the tangent norm is already clipped to MAX_NORM)
// ---------------------------------------------------------------------------
__global__ __launch_bounds__(256)
void hyper_epilogue(const float* __restrict__ Y, __bf16* __restrict__ An, int H) {
  __shared__ float sm[256];
  const size_t row = blockIdx.x;
  const int t = threadIdx.x;
  const int Q = H >> 8;           // elements per thread (H=1024 -> 4)
  float y[4];
  float local = 0.0f;
  for (int q = 0; q < Q; q++) {
    const int j = t + (q << 8);
    y[q] = Y[row * (size_t)H + j];
    if (j > 0) local += y[q] * y[q];
  }
  const float S  = block_reduce256(local, sm);
  const float n  = sqrtf(S);
  const float sc = fminf(1.0f, MAXN / fmaxf(n, EPSF));
  const float m  = fminf(n, MAXN);
  const float f1 = sinh_div(m) * sc;
  const float d  = acoshf(fmaxf(coshf(m), 1.0f + EPSF));
  const float f  = div_sinh(d) * f1;
  for (int q = 0; q < Q; q++) {
    const int j = t + (q << 8);
    const float a = (j == 0) ? 0.0f : f * y[q];
    An[row * (size_t)H + j] = (__bf16)a;
  }
}

// ---------------------------------------------------------------------------
// Final stage: Y2(N,256) tangent output + z_h -> z_next; also copies z_h out.
// ---------------------------------------------------------------------------
__global__ __launch_bounds__(256)
void lorentz_final(const float* __restrict__ Y2, const float* __restrict__ zh,
                   const float* __restrict__ alphaP, const float* __restrict__ stepP,
                   float* __restrict__ out, int N) {
  __shared__ float sm[256];
  const size_t row = blockIdx.x;
  const int j = threadIdx.x;
  const float y = Y2[row * 256 + j];
  const float z = zh[row * 256 + j];
  const float step = 1.0f / (1.0f + expf(-stepP[0]));
  const float a_s  = 1.0f / (1.0f + expf(-alphaP[0]));

  // v_transformed = logmap0(projx(safe_expmap0(y)))
  const float S1  = block_reduce256((j == 0) ? 0.0f : y * y, sm);
  const float n1  = sqrtf(S1);
  const float sc1 = fminf(1.0f, MAXN / fmaxf(n1, EPSF));
  const float m1  = fminf(n1, MAXN);
  const float d1  = acoshf(fmaxf(coshf(m1), 1.0f + EPSF));
  const float vt  = (j == 0) ? 0.0f : div_sinh(d1) * sinh_div(m1) * sc1 * y;
  const float vf  = step * vt;                        // v_final, vf[0]=0

  // z_update = projx(safe_expmap(z_h, v_final))
  const float S2  = block_reduce256((j == 0) ? (-vf * vf) : (vf * vf), sm);
  const float n2  = sqrtf(fmaxf(S2, 0.0f));
  const float sc2 = fminf(1.0f, MAXN / fmaxf(n2, EPSF));
  const float m2  = fminf(n2, MAXN);
  float zu = coshf(m2) * z + sinh_div(m2) * sc2 * vf;
  const float S3 = block_reduce256((j == 0) ? 0.0f : zu * zu, sm);
  if (j == 0) zu = sqrtf(1.0f + S3);

  // tangent_direction = logmap(z_h, z_update); scaled by (1 - sigmoid(alpha))
  const float S4 = block_reduce256((j == 0) ? (-z * zu) : (z * zu), sm);
  const float ip = fminf(S4, -(1.0f + EPSF));
  const float d3 = acoshf(-ip);
  const float w  = (1.0f - a_s) * (div_sinh(d3) * (zu + ip * z));

  // z_next = projx(safe_expmap(z_h, w))
  const float S5  = block_reduce256((j == 0) ? (-w * w) : (w * w), sm);
  const float n5  = sqrtf(fmaxf(S5, 0.0f));
  const float sc5 = fminf(1.0f, MAXN / fmaxf(n5, EPSF));
  const float m5  = fminf(n5, MAXN);
  float zn = coshf(m5) * z + sinh_div(m5) * sc5 * w;
  const float S6 = block_reduce256((j == 0) ? 0.0f : zn * zn, sm);
  if (j == 0) zn = sqrtf(1.0f + S6);

  out[row * 256 + j] = zn;
  out[(size_t)N * 256 + row * 256 + j] = z;   // second tuple element: z_h
}

// ---------------------------------------------------------------------------
// Launch
// ---------------------------------------------------------------------------
extern "C" void kernel_launch(void* const* d_in, const int* in_sizes, int n_in,
                              void* d_out, int out_size, void* d_ws, size_t ws_size,
                              hipStream_t stream) {
  (void)n_in; (void)out_size; (void)ws_size;
  const float* zh = (const float*)d_in[0];
  const float* w0 = (const float*)d_in[1];
  const float* b0 = (const float*)d_in[2];
  const float* w1 = (const float*)d_in[3];
  const float* b1 = (const float*)d_in[4];
  const float* w2 = (const float*)d_in[5];
  const float* b2 = (const float*)d_in[6];
  const float* al = (const float*)d_in[7];
  const float* st = (const float*)d_in[8];

  const int D = 256;
  const int N = in_sizes[0] / D;        // 32768
  const int H = in_sizes[1] / D;        // w0 is (H, D) -> 1024

  // workspace: bufA/bufB (bf16 activations), bufY (f32 GEMM out), bf16 weights
  char* ws = (char*)d_ws;
  size_t off = 0;
  __bf16* bufA = (__bf16*)(ws + off); off += (size_t)N * H * sizeof(__bf16);
  __bf16* bufB = (__bf16*)(ws + off); off += (size_t)N * H * sizeof(__bf16);
  float*  bufY = (float*)(ws + off);  off += (size_t)N * H * sizeof(float);
  __bf16* w0b  = (__bf16*)(ws + off); off += (size_t)H * D * sizeof(__bf16);
  __bf16* w1b  = (__bf16*)(ws + off); off += (size_t)H * H * sizeof(__bf16);
  __bf16* w2b  = (__bf16*)(ws + off);

  f32_to_bf16<<<(H * D + 255) / 256, 256, 0, stream>>>(w0, w0b, H * D);
  f32_to_bf16<<<(H * H + 255) / 256, 256, 0, stream>>>(w1, w1b, H * H);
  f32_to_bf16<<<(D * H + 255) / 256, 256, 0, stream>>>(w2, w2b, D * H);

  prep_rows<<<N, 256, 0, stream>>>(zh, bufA);

  gemm_bf16_wmma<<<dim3(N / BM, H / BN), 256, 0, stream>>>(bufA, w0b, b0, bufY, D, H);
  hyper_epilogue<<<N, 256, 0, stream>>>(bufY, bufB, H);

  gemm_bf16_wmma<<<dim3(N / BM, H / BN), 256, 0, stream>>>(bufB, w1b, b1, bufY, H, H);
  hyper_epilogue<<<N, 256, 0, stream>>>(bufY, bufA, H);

  gemm_bf16_wmma<<<dim3(N / BM, D / BN), 256, 0, stream>>>(bufA, w2b, b2, bufY, H, D);

  lorentz_final<<<N, 256, 0, stream>>>(bufY, zh, al, st, (float*)d_out, N);
}
